// MambaWrapper_38319698215502
// MI455X (gfx1250) — matrix operand order repaired and checked by reading, not compiled, not run on hardware
//
#include <hip/hip_runtime.h>
#include <stdint.h>

// ---------------------------------------------------------------------------
// Mamba forward for MI455X (gfx1250).
// Dense einsums: V_WMMA_F32_16X16X32_BF16, 64x64 block tiles, 4 WMMA/wave/K-step,
// ping-pong LDS pipeline: A tiles prefetched by the Tensor Data Mover
// (tensor_load_to_lds, TENSORcnt) one K-step ahead; B tiles staged with wide
// f32 loads + on-the-fly bf16 convert. Selective scan: register recurrence.
// ---------------------------------------------------------------------------

typedef __attribute__((ext_vector_type(16))) __bf16 v16bf;
typedef __attribute__((ext_vector_type(8)))  __bf16 v8bf;
typedef __attribute__((ext_vector_type(8)))  float  v8f;
typedef __attribute__((ext_vector_type(4)))  unsigned int u32x4;
typedef __attribute__((ext_vector_type(8)))  int    i32x8;
typedef __attribute__((ext_vector_type(4)))  int    i32x4;

#define D_MODEL 768
#define D_INNER 1536
#define D_STATE 16
#define D_CONV  4
#define DT_RANK 48
#define N_LAYER 4
#define SEQ_L   1024
#define VOCAB   50257
#define PROJ_W  (DT_RANK + 2 * D_STATE)   // 80

#define BM 64
#define BN 64
#define BK 32
#define BN_PAD 8   // BsT row stride = 72 halves = 144B (16B aligned)

// float -> bf16, round-to-nearest-even
__device__ __forceinline__ __bf16 f2bf(float f) {
  uint32_t u; __builtin_memcpy(&u, &f, 4);
  uint32_t r = u + 0x7FFFu + ((u >> 16) & 1u);
  uint16_t hv = (uint16_t)(r >> 16);
  __bf16 out; __builtin_memcpy(&out, &hv, 2);
  return out;
}

__device__ __forceinline__ v16bf cat16(v8bf lo, v8bf hi) {
  return __builtin_shufflevector(lo, hi, 0, 1, 2, 3, 4, 5, 6, 7,
                                 8, 9, 10, 11, 12, 13, 14, 15);
}

// Stage one 64x32 A tile (TDM, wave 0) + 32x64 B tile (wide loads, transposed)
__device__ __forceinline__ void stage_tile(
    const __bf16* __restrict__ A, const float* __restrict__ B,
    int M, int N, int K, int m0, int n0, int kt, int tid, int wave,
    __bf16 (*As)[BK], __bf16 (*BsT)[BN + BN_PAD]) {
  const __bf16 ZB = f2bf(0.f);
  const bool full = (kt + BK <= K) && (n0 + BN <= N);
  if (full) {
    // ---- A tile via Tensor Data Mover: 2D D#, 64 rows x 32 cols of 2B ----
    if (wave == 0) {
      uint64_t ga = (uint64_t)(uintptr_t)A +
                    (((uint64_t)(size_t)m0 * (uint64_t)K + (uint64_t)kt) << 1);
      uint32_t lds = (uint32_t)(uintptr_t)&As[0][0];  // addr[31:0] = LDS offset
      u32x4 g0;
      g0[0] = 1u;                                    // count=1, user mode
      g0[1] = lds;                                   // lds_addr
      g0[2] = (uint32_t)ga;                          // global_addr[31:0]
      g0[3] = (uint32_t)((ga >> 32) & 0x1FFFFFFu)    // global_addr[56:32]
            | (2u << 30);                            // type=2 ("image")
      i32x8 g1;
      g1[0] = (int)(1u << 16);                       // data_size=1 -> 2 bytes
      g1[1] = (int)(((uint32_t)K & 0xFFFFu) << 16);  // tensor_dim0[15:0]
      g1[2] = (int)((((uint32_t)K >> 16) & 0xFFFFu)
            | (((uint32_t)M & 0xFFFFu) << 16));      // dim0 hi | dim1 lo
      g1[3] = (int)((((uint32_t)M >> 16) & 0xFFFFu)
            | ((uint32_t)BK << 16));                 // dim1 hi | tile_dim0=32
      g1[4] = BM;                                    // tile_dim1=64, tile_dim2=0
      g1[5] = K;                                     // tensor_dim0_stride lo
      g1[6] = 0;
      g1[7] = 0;
      i32x4 gz = {0, 0, 0, 0};
#if defined(__clang_major__) && __clang_major__ >= 23
      i32x8 gz8 = {0, 0, 0, 0, 0, 0, 0, 0};
      __builtin_amdgcn_tensor_load_to_lds(g0, g1, gz, gz, gz8, 0);
#else
      __builtin_amdgcn_tensor_load_to_lds(g0, g1, gz, gz, 0);
#endif
    }
    // ---- B tile: wide f32 loads, convert, transpose-store into BsT ----
    const int n  = tid >> 1;            // 0..63
    const int ks = (tid & 1) * 16;      // 0 or 16
    const float* src = B + (size_t)(n0 + n) * K + (kt + ks);
    float4 f0 = *(const float4*)(src + 0);
    float4 f1 = *(const float4*)(src + 4);
    float4 f2 = *(const float4*)(src + 8);
    float4 f3 = *(const float4*)(src + 12);
    BsT[ks + 0][n]  = f2bf(f0.x); BsT[ks + 1][n]  = f2bf(f0.y);
    BsT[ks + 2][n]  = f2bf(f0.z); BsT[ks + 3][n]  = f2bf(f0.w);
    BsT[ks + 4][n]  = f2bf(f1.x); BsT[ks + 5][n]  = f2bf(f1.y);
    BsT[ks + 6][n]  = f2bf(f1.z); BsT[ks + 7][n]  = f2bf(f1.w);
    BsT[ks + 8][n]  = f2bf(f2.x); BsT[ks + 9][n]  = f2bf(f2.y);
    BsT[ks + 10][n] = f2bf(f2.z); BsT[ks + 11][n] = f2bf(f2.w);
    BsT[ks + 12][n] = f2bf(f3.x); BsT[ks + 13][n] = f2bf(f3.y);
    BsT[ks + 14][n] = f2bf(f3.z); BsT[ks + 15][n] = f2bf(f3.w);
  } else {
    // ---- edge path: clamped (always-legal) loads + select, branch-free ----
    for (int t = tid; t < BM * BK; t += 128) {
      int m = t >> 5, kk = t & 31;
      int gk = kt + kk;
      int gkc = (gk < K) ? gk : (K - 1);
      __bf16 v = A[(size_t)(m0 + m) * K + gkc];   // m0+m < M always (M%64==0)
      As[m][kk] = (gk < K) ? v : ZB;
    }
    for (int t = tid; t < BK * BN; t += 128) {
      int n = t & 63, kk = t >> 6;
      int gn = n0 + n, gk = kt + kk;
      int gnc = (gn < N) ? gn : (N - 1);
      int gkc = (gk < K) ? gk : (K - 1);
      float v = B[(size_t)gnc * K + gkc];
      BsT[kk][n] = (gn < N && gk < K) ? f2bf(v) : ZB;
    }
  }
}

// ---------------------------------------------------------------------------
// GEMM-NT:  C[m,n] = sum_k A[m,k]*B[n,k]  (+ optional Cin residual)
// A: bf16 MxK row-major (activations).  B: f32 NxK row-major (weights).
// Block = 128 threads (4 waves); tile 64(M) x 64(N); K-step 32, ping-pong LDS.
// ---------------------------------------------------------------------------
__global__ __launch_bounds__(128)
void gemm_bf16_nt(const __bf16* __restrict__ A, const float* __restrict__ B,
                  float* __restrict__ C, const float* __restrict__ Cin,
                  int M, int N, int K) {
  __shared__ __align__(16) __bf16 As[2][BM][BK];
  __shared__ __align__(16) __bf16 BsT[2][BK][BN + BN_PAD];

  const int tid  = threadIdx.x;
  const int lane = tid & 31;
  const int wave = tid >> 5;
  const int m0 = blockIdx.y * BM;
  const int n0 = blockIdx.x * BN;

  v8f acc[4] = {};

  const int nsteps = (K + BK - 1) / BK;

  // prologue: stage tile 0
  stage_tile(A, B, M, N, K, m0, n0, 0, tid, wave, As[0], BsT[0]);
  if (wave == 0) __builtin_amdgcn_s_wait_tensorcnt(0);
  __syncthreads();

  for (int s = 0; s < nsteps; ++s) {
    const int cur = s & 1;
    const int nxt = cur ^ 1;
    // prefetch next tile into the alternate buffer (TDM + wide B loads)
    if (s + 1 < nsteps)
      stage_tile(A, B, M, N, K, m0, n0, (s + 1) * BK, tid, wave, As[nxt], BsT[nxt]);

    // B fragment: lane&15 = N; K runs {kh..kh+7} and {16+kh..16+kh+7} (same as A)
    const int bn  = wave * 16 + (lane & 15);
    const int bkh = (lane >> 4) * 8;
    v16bf bfrag;
#pragma unroll
    for (int e = 0; e < 8; ++e) {
      bfrag[e]     = BsT[cur][bkh + e][bn];
      bfrag[8 + e] = BsT[cur][16 + bkh + e][bn];
    }

    // A fragment: lane&15 = M; K runs {kh..kh+7} and {16+kh..16+kh+7} contig
    const int mr = lane & 15;
    const int kh = (lane >> 4) * 8;
#pragma unroll
    for (int ms = 0; ms < 4; ++ms) {
      const __bf16* ap = &As[cur][ms * 16 + mr][0] + kh;
      v16bf afrag = cat16(*(const v8bf*)ap, *(const v8bf*)(ap + 16));
      acc[ms] = __builtin_amdgcn_wmma_f32_16x16x32_bf16(
          false, afrag, false, bfrag, (short)0, acc[ms], false, false);
    }
    // wave 0 owns all outstanding TDM ops; publish next buffer to the block
    if (wave == 0) __builtin_amdgcn_s_wait_tensorcnt(0);
    __syncthreads();
  }

  // C/D layout: lane 0-15 -> N=lane, M=r; lanes 16-31 -> N=lane-16, M=8+r
  const int col = n0 + wave * 16 + (lane & 15);
  const int mo  = (lane >> 4) * 8;
#pragma unroll
  for (int ms = 0; ms < 4; ++ms) {
#pragma unroll
    for (int r = 0; r < 8; ++r) {
      int row = m0 + ms * 16 + mo + r;
      if (col < N) {   // row < M always: M % 64 == 0 for all call sites
        float v = acc[ms][r];
        if (Cin) v += Cin[(size_t)row * N + col];
        C[(size_t)row * N + col] = v;
      }
    }
  }
}

// ---------------------------------------------------------------------------
// Elementwise / small kernels
// ---------------------------------------------------------------------------
__global__ void embed_gather(const int* __restrict__ x, const float* __restrict__ embed,
                             float* __restrict__ h) {
  int idx = blockIdx.x * blockDim.x + threadIdx.x;
  if (idx >= SEQ_L * D_MODEL) return;
  int l = idx / D_MODEL, d = idx - l * D_MODEL;
  h[idx] = embed[(size_t)x[l] * D_MODEL + d];
}

__global__ void rmsnorm_bf16(const float* __restrict__ h, const float* __restrict__ w,
                             __bf16* __restrict__ out) {
  __shared__ float red[256];
  const int row = blockIdx.x;
  const float* hr = h + (size_t)row * D_MODEL;
  float s = 0.f;
  for (int t = threadIdx.x; t < D_MODEL; t += 256) { float v = hr[t]; s += v * v; }
  red[threadIdx.x] = s;
  __syncthreads();
  for (int off = 128; off > 0; off >>= 1) {
    if (threadIdx.x < off) red[threadIdx.x] += red[threadIdx.x + off];
    __syncthreads();
  }
  float rs = rsqrtf(red[0] / (float)D_MODEL + 1e-5f);
  for (int t = threadIdx.x; t < D_MODEL; t += 256)
    out[(size_t)row * D_MODEL + t] = f2bf(hr[t] * rs * w[t]);
}

// causal depthwise conv (D_CONV=4) + SiLU
__global__ void conv_silu(const float* __restrict__ xz, const float* __restrict__ cw,
                          const float* __restrict__ cb, float* __restrict__ xc,
                          __bf16* __restrict__ xcbf) {
  int idx = blockIdx.x * blockDim.x + threadIdx.x;
  if (idx >= SEQ_L * D_INNER) return;
  int l = idx / D_INNER, i = idx - l * D_INNER;
  float acc = cb[i];
#pragma unroll
  for (int k = 0; k < D_CONV; ++k) {
    int ls = l + k - (D_CONV - 1);
    if (ls >= 0) acc += xz[(size_t)ls * (2 * D_INNER) + i] * cw[i * D_CONV + k];
  }
  float s = acc / (1.f + __expf(-acc));
  xc[idx]   = s;
  xcbf[idx] = f2bf(s);
}

__global__ void split_dt(const float* __restrict__ proj, __bf16* __restrict__ dtbf) {
  int idx = blockIdx.x * blockDim.x + threadIdx.x;
  if (idx >= SEQ_L * DT_RANK) return;
  int l = idx / DT_RANK, r = idx - l * DT_RANK;
  dtbf[idx] = f2bf(proj[(size_t)l * PROJ_W + r]);
}

__global__ void bias_softplus(float* __restrict__ dt, const float* __restrict__ b) {
  int idx = blockIdx.x * blockDim.x + threadIdx.x;
  if (idx >= SEQ_L * D_INNER) return;
  int i = idx % D_INNER;
  float v = dt[idx] + b[i];
  dt[idx] = (v > 20.f) ? v : log1pf(__expf(v));
}

// Selective scan, fused dA/dBx/state-update/C-contraction/D-skip/SiLU(z) gate
__global__ void ssm_scan(const float* __restrict__ dt, const float* __restrict__ proj,
                         const float* __restrict__ xc, const float* __restrict__ xz,
                         const float* __restrict__ A_log, const float* __restrict__ Dp,
                         __bf16* __restrict__ ybf) {
  int i = blockIdx.x * blockDim.x + threadIdx.x;
  if (i >= D_INNER) return;
  float A[D_STATE];
#pragma unroll
  for (int s = 0; s < D_STATE; ++s) A[s] = -__expf(A_log[(size_t)i * D_STATE + s]);
  const float Dv = Dp[i];
  float hst[D_STATE];
#pragma unroll
  for (int s = 0; s < D_STATE; ++s) hst[s] = 0.f;

  for (int l = 0; l < SEQ_L; ++l) {
    float dtv = dt[(size_t)l * D_INNER + i];
    float x   = xc[(size_t)l * D_INNER + i];
    const float* pr = proj + (size_t)l * PROJ_W;
    float acc = 0.f;
#pragma unroll
    for (int s = 0; s < D_STATE; ++s) {
      float da = __expf(dtv * A[s]);
      hst[s] = da * hst[s] + (dtv * pr[DT_RANK + s]) * x;
      acc += hst[s] * pr[DT_RANK + D_STATE + s];
    }
    float y = acc + x * Dv;
    float z = xz[(size_t)l * (2 * D_INNER) + D_INNER + i];
    float sz = z / (1.f + __expf(-z));
    ybf[(size_t)l * D_INNER + i] = f2bf(y * sz);
  }
}

__global__ void write_scalar(float* __restrict__ out, int idx) {
  out[idx] = 0.f;
}

// ---------------------------------------------------------------------------
// Launch
// ---------------------------------------------------------------------------
extern "C" void kernel_launch(void* const* d_in, const int* in_sizes, int n_in,
                              void* d_out, int out_size, void* d_ws, size_t ws_size,
                              hipStream_t stream) {
  (void)in_sizes; (void)n_in; (void)ws_size;
  const int*   x        = (const int*)  d_in[0];
  const float* embed    = (const float*)d_in[1];
  const float* norm_w   = (const float*)d_in[2];
  const float* in_w     = (const float*)d_in[3];
  const float* conv_w   = (const float*)d_in[4];
  const float* conv_b   = (const float*)d_in[5];
  const float* xproj_w  = (const float*)d_in[6];
  const float* dtproj_w = (const float*)d_in[7];
  const float* dtproj_b = (const float*)d_in[8];
  const float* A_log    = (const float*)d_in[9];
  const float* Dp       = (const float*)d_in[10];
  const float* out_w    = (const float*)d_in[11];
  const float* normf_w  = (const float*)d_in[12];
  float* logits = (float*)d_out;

  char* ws = (char*)d_ws;
  size_t off = 0;
  auto alloc = [&](size_t bytes) -> char* {
    char* p = ws + off;
    off = (off + bytes + 255) & ~(size_t)255;
    return p;
  };
  float*  h     = (float*) alloc((size_t)SEQ_L * D_MODEL * 4);
  __bf16* hnbf  = (__bf16*)alloc((size_t)SEQ_L * D_MODEL * 2);
  float*  xzbuf = (float*) alloc((size_t)SEQ_L * 2 * D_INNER * 4);
  float*  xcbuf = (float*) alloc((size_t)SEQ_L * D_INNER * 4);
  __bf16* xcbf  = (__bf16*)alloc((size_t)SEQ_L * D_INNER * 2);
  float*  projb = (float*) alloc((size_t)SEQ_L * PROJ_W * 4);
  __bf16* dtbf  = (__bf16*)alloc((size_t)SEQ_L * DT_RANK * 2);
  float*  dtful = (float*) alloc((size_t)SEQ_L * D_INNER * 4);
  __bf16* ybf   = (__bf16*)alloc((size_t)SEQ_L * D_INNER * 2);

  const dim3 b256(256);
  const int gy = SEQ_L / BM;   // 16
  embed_gather<<<(SEQ_L * D_MODEL + 255) / 256, b256, 0, stream>>>(x, embed, h);

  for (int i = 0; i < N_LAYER; ++i) {
    rmsnorm_bf16<<<SEQ_L, b256, 0, stream>>>(h, norm_w + (size_t)i * D_MODEL, hnbf);
    gemm_bf16_nt<<<dim3((2 * D_INNER + BN - 1) / BN, gy), 128, 0, stream>>>(
        hnbf, in_w + (size_t)i * 2 * D_INNER * D_MODEL, xzbuf, nullptr,
        SEQ_L, 2 * D_INNER, D_MODEL);
    conv_silu<<<(SEQ_L * D_INNER + 255) / 256, b256, 0, stream>>>(
        xzbuf, conv_w + (size_t)i * D_INNER * D_CONV, conv_b + (size_t)i * D_INNER,
        xcbuf, xcbf);
    gemm_bf16_nt<<<dim3((PROJ_W + BN - 1) / BN, gy), 128, 0, stream>>>(
        xcbf, xproj_w + (size_t)i * PROJ_W * D_INNER, projb, nullptr,
        SEQ_L, PROJ_W, D_INNER);
    split_dt<<<(SEQ_L * DT_RANK + 255) / 256, b256, 0, stream>>>(projb, dtbf);
    gemm_bf16_nt<<<dim3((D_INNER + BN - 1) / BN, gy), 128, 0, stream>>>(
        dtbf, dtproj_w + (size_t)i * D_INNER * DT_RANK, dtful, nullptr,
        SEQ_L, D_INNER, DT_RANK);
    bias_softplus<<<(SEQ_L * D_INNER + 255) / 256, b256, 0, stream>>>(
        dtful, dtproj_b + (size_t)i * D_INNER);
    ssm_scan<<<(D_INNER + 255) / 256, b256, 0, stream>>>(
        dtful, projb, xcbuf, xzbuf,
        A_log + (size_t)i * D_INNER * D_STATE, Dp + (size_t)i * D_INNER, ybf);
    gemm_bf16_nt<<<dim3((D_MODEL + BN - 1) / BN, gy), 128, 0, stream>>>(
        ybf, out_w + (size_t)i * D_MODEL * D_INNER, h, h,
        SEQ_L, D_MODEL, D_INNER);
  }

  rmsnorm_bf16<<<SEQ_L, b256, 0, stream>>>(h, normf_w, hnbf);
  gemm_bf16_nt<<<dim3((VOCAB + BN - 1) / BN, gy), 128, 0, stream>>>(
      hnbf, embed, logits, nullptr, SEQ_L, VOCAB, D_MODEL);
  write_scalar<<<1, 1, 0, stream>>>(logits, out_size - 1);
}
